// MultiHeadAttention_18073222381989
// MI455X (gfx1250) — matrix operand order repaired and checked by reading, not compiled, not run on hardware
//
#include <hip/hip_runtime.h>
#include <hip/hip_bf16.h>

// ---------------------------------------------------------------------------
// MultiHeadAttention forward for MI455X (gfx1250, wave32, WMMA).
// Shapes: B=16, T=1024, D_IN=512, H=8, D_K=32, DH=64.
// outputs (concat in d_out): output [h,b,t,dh] f32 (8.4M), attn [h,b,t,t] f32 (134M)
// Roofline: ~34 GFLOP vs 537 MB streamed attn write -> HBM-store bound;
// all GEMMs in bf16 WMMA (f32 accum), attn written once with NT hint.
// ---------------------------------------------------------------------------

typedef __bf16 bf16_t;
typedef __attribute__((ext_vector_type(16))) __bf16 v16bf;
typedef __attribute__((ext_vector_type(8)))  float  v8f;
typedef __attribute__((ext_vector_type(4)))  float  f4v;

#define NHEAD 8
#define DK    32
#define DIN   512
#define BATCH 16
#define SEQ   1024
#define DH    64
#define EPROJ 256   // NHEAD*DK

// Load 16 bf16 (two 16B chunks at independent addresses) into a v16bf.
static __device__ __forceinline__ v16bf load_pair(const bf16_t* p0, const bf16_t* p1) {
  union { uint4 u[2]; v16bf v; } r;
  r.u[0] = *(const uint4*)p0;
  r.u[1] = *(const uint4*)p1;
  return r.v;
}

// ---------------------------------------------------------------------------
// Kernel 0: weights f32 -> bf16
// ---------------------------------------------------------------------------
__global__ void mha_cvt_w(const float* __restrict__ Wq, const float* __restrict__ Wk,
                          bf16_t* __restrict__ Wqb, bf16_t* __restrict__ Wkb) {
  int i = blockIdx.x * blockDim.x + threadIdx.x;   // 0 .. 131071
  if (i < EPROJ * DIN) {
    Wqb[i] = (bf16_t)Wq[i];
    Wkb[i] = (bf16_t)Wk[i];
  }
}

// ---------------------------------------------------------------------------
// Kernel 1: Vt[b][h][n(64)][t(1024)] = bf16(v[b][t][h*64+n])   (write-coalesced)
// ---------------------------------------------------------------------------
__global__ void mha_build_vt(const float* __restrict__ v, bf16_t* __restrict__ Vt) {
  int idx = blockIdx.x * blockDim.x + threadIdx.x;   // 0 .. 8388607
  int t = idx & (SEQ - 1);
  int n = (idx >> 10) & (DH - 1);
  int h = (idx >> 16) & (NHEAD - 1);
  int b = idx >> 19;
  Vt[idx] = (bf16_t)v[(size_t)(b * SEQ + t) * DIN + h * DH + n];
}

// ---------------------------------------------------------------------------
// Kernel 2: projections Q = v*Wq^T + bq, K = v*Wk^T + bk, stored bf16 [b,h,t,32]
// One wave per 16x16 output tile; K=512 in 16 WMMA steps; A converted on the fly.
// ---------------------------------------------------------------------------
__global__ __launch_bounds__(32) void mha_proj(
    const float*  __restrict__ v,
    const bf16_t* __restrict__ Wqb, const bf16_t* __restrict__ Wkb,
    const float*  __restrict__ bq,  const float*  __restrict__ bk,
    bf16_t* __restrict__ Qh, bf16_t* __restrict__ Kh) {
  const int x = blockIdx.x;              // token tile: 0..1023
  const int y = blockIdx.y;              // e tile:     0..15
  const int L = threadIdx.x;
  const int half = L >> 4;               // 0: lanes 0-15, 1: lanes 16-31
  const int lr = L & 15;

  const float*  vrow = v + (size_t)(x * 16 + lr) * DIN;          // A row (M = lr)
  const int     e    = y * 16 + lr;                              // B col (N = lr)
  const bf16_t* wqr  = Wqb + (size_t)e * DIN;
  const bf16_t* wkr  = Wkb + (size_t)e * DIN;

  v8f cq = {0,0,0,0,0,0,0,0};
  v8f ck = {0,0,0,0,0,0,0,0};

  #pragma unroll
  for (int kb = 0; kb < 16; kb++) {
    // A: 16x32 f32 -> bf16. lane<16: K {0..7,16..23}; lane>=16: {8..15,24..31}
    const float* p = vrow + kb * 32 + half * 8;
    float4 x0 = *(const float4*)(p);
    float4 x1 = *(const float4*)(p + 4);
    float4 x2 = *(const float4*)(p + 16);
    float4 x3 = *(const float4*)(p + 20);
    v16bf a;
    a[0]=(bf16_t)x0.x; a[1]=(bf16_t)x0.y; a[2]=(bf16_t)x0.z; a[3]=(bf16_t)x0.w;
    a[4]=(bf16_t)x1.x; a[5]=(bf16_t)x1.y; a[6]=(bf16_t)x1.z; a[7]=(bf16_t)x1.w;
    a[8]=(bf16_t)x2.x; a[9]=(bf16_t)x2.y; a[10]=(bf16_t)x2.z; a[11]=(bf16_t)x2.w;
    a[12]=(bf16_t)x3.x; a[13]=(bf16_t)x3.y; a[14]=(bf16_t)x3.z; a[15]=(bf16_t)x3.w;

    // B: same K order as A; lane<16: K {0..7,16..23}; lane>=16: {8..15,24..31}
    const bf16_t* pq = wqr + kb * 32 + half * 8;
    const bf16_t* pk = wkr + kb * 32 + half * 8;
    v16bf bqm = load_pair(pq, pq + 16);
    v16bf bkm = load_pair(pk, pk + 16);

    cq = __builtin_amdgcn_wmma_f32_16x16x32_bf16(false, a, false, bqm, (short)0, cq, false, false);
    ck = __builtin_amdgcn_wmma_f32_16x16x32_bf16(false, a, false, bkm, (short)0, ck, false, false);
  }

  const float bqv = bq[e];
  const float bkv = bk[e];
  const int hh = e >> 5;        // head
  const int dk = e & 31;        // within-head dim

  #pragma unroll
  for (int j = 0; j < 8; j++) {
    int g  = x * 16 + j + half * 8;    // global token
    int bb = g >> 10;
    int tt = g & (SEQ - 1);
    size_t idx = ((size_t)((bb * NHEAD + hh) * SEQ + tt)) * DK + dk;
    Qh[idx] = (bf16_t)(cq[j] + bqv);   // re-read soon by attention -> keep RT (L2-resident)
    Kh[idx] = (bf16_t)(ck[j] + bkv);
  }
}

// ---------------------------------------------------------------------------
// Kernel 3: fused attention. Grid (64 q-tiles, 128 b*h), 8 waves/block.
// Wave w owns k-columns [w*128, w*128+128): 8 score WMMAs + 16 PV WMMAs.
// ---------------------------------------------------------------------------
union ShMem {
  bf16_t p[8][16][128];   // bf16 probs, wave-private strips (A-operand staging)
  float  o[8][16][64];    // O partials (reuses same space after P consumed)
};

__global__ __launch_bounds__(256) void mha_attn(
    const bf16_t* __restrict__ Qh, const bf16_t* __restrict__ Kh,
    const bf16_t* __restrict__ Vt, const unsigned char* __restrict__ mask,
    float* __restrict__ out, float* __restrict__ attn) {
  __shared__ ShMem sh;
  __shared__ float redmax[8][16];
  __shared__ float redsum[8][16];

  const int qb = blockIdx.x;
  const int h  = blockIdx.y & (NHEAD - 1);
  const int b  = blockIdx.y >> 3;
  const int w  = threadIdx.x >> 5;
  const int L  = threadIdx.x & 31;
  const int half = L >> 4;
  const int lr   = L & 15;

  const bf16_t* Qbh = Qh + (size_t)(b * NHEAD + h) * SEQ * DK;
  const bf16_t* Kbh = Kh + (size_t)(b * NHEAD + h) * SEQ * DK;
  const bf16_t* Vbh = Vt + (size_t)(b * NHEAD + h) * DH * SEQ;

  // ---- Q as A operand (16 rows x K=32 == d_k) ----
  const bf16_t* qrow = Qbh + (size_t)(qb * 16 + lr) * DK + half * 8;
  const v16bf a = load_pair(qrow, qrow + 16);

  // ---- scores: 8 tiles, one WMMA each ----
  const float inv_t = 0.17677669529663687f;   // 1/sqrt(32)
  v8f c[8];
  #pragma unroll
  for (int i = 0; i < 8; i++) {
    const int k0 = w * 128 + i * 16;
    const bf16_t* krow = Kbh + (size_t)(k0 + lr) * DK + half * 8;  // B: lane=N(col t)
    v16bf bt = load_pair(krow, krow + 16);
    v8f z = {0,0,0,0,0,0,0,0};
    c[i] = __builtin_amdgcn_wmma_f32_16x16x32_bf16(false, a, false, bt, (short)0, z, false, false);
    const bool m = mask[b * SEQ + k0 + lr] != 0;   // per-column mask
    #pragma unroll
    for (int j = 0; j < 8; j++) c[i][j] = m ? -1000.0f : c[i][j] * inv_t;
  }

  // ---- softmax row stats (row = j + half*8) ----
  float fm[8], rs[8];
  #pragma unroll
  for (int j = 0; j < 8; j++) {
    float m = c[0][j];
    #pragma unroll
    for (int i = 1; i < 8; i++) m = fmaxf(m, c[i][j]);
    #pragma unroll
    for (int s = 1; s < 16; s <<= 1) m = fmaxf(m, __shfl_xor(m, s, 32));
    fm[j] = m;
  }
  if (lr == 0) {
    #pragma unroll
    for (int j = 0; j < 8; j++) redmax[w][half * 8 + j] = fm[j];
  }
  __syncthreads();
  #pragma unroll
  for (int j = 0; j < 8; j++) {
    float m = -3.0e38f;
    #pragma unroll
    for (int ww = 0; ww < 8; ww++) m = fmaxf(m, redmax[ww][half * 8 + j]);
    fm[j] = m;
  }
  #pragma unroll
  for (int j = 0; j < 8; j++) {
    float s = 0.0f;
    #pragma unroll
    for (int i = 0; i < 8; i++) { float e = __expf(c[i][j] - fm[j]); c[i][j] = e; s += e; }
    #pragma unroll
    for (int m = 1; m < 16; m <<= 1) s += __shfl_xor(s, m, 32);
    rs[j] = s;
  }
  if (lr == 0) {
    #pragma unroll
    for (int j = 0; j < 8; j++) redsum[w][half * 8 + j] = rs[j];
  }
  __syncthreads();
  #pragma unroll
  for (int j = 0; j < 8; j++) {
    float s = 0.0f;
    #pragma unroll
    for (int ww = 0; ww < 8; ww++) s += redsum[ww][half * 8 + j];
    rs[j] = 1.0f / s;
  }

  // ---- normalize, stream attn to HBM non-temporally (write-once, 537 MB),
  //      stage bf16 P in wave-private LDS for the PV GEMM ----
  const size_t abase = ((size_t)(h * BATCH + b) * SEQ + qb * 16) * (size_t)SEQ;
  #pragma unroll
  for (int i = 0; i < 8; i++) {
    const int col = w * 128 + i * 16 + lr;
    #pragma unroll
    for (int j = 0; j < 8; j++) {
      const int row = j + half * 8;
      const float pv = c[i][j] * rs[j];
      __builtin_nontemporal_store(pv, attn + abase + (size_t)row * SEQ + col);
      sh.p[w][row][i * 16 + lr] = (bf16_t)pv;
    }
  }

  // ---- O partial = P(16x128) * Vt(128x64): 4 K-chunks x 4 N-tiles ----
  v8f z8 = {0,0,0,0,0,0,0,0};
  v8f acc[4] = {z8, z8, z8, z8};
  #pragma unroll
  for (int kk = 0; kk < 4; kk++) {
    const bf16_t* ap = &sh.p[w][lr][kk * 32 + half * 8];       // A from LDS strip
    v16bf a2 = load_pair(ap, ap + 16);
    #pragma unroll
    for (int nt = 0; nt < 4; nt++) {
      const bf16_t* vp = Vbh + (size_t)(nt * 16 + lr) * SEQ + (w * 128 + kk * 32 + half * 8);
      v16bf b2 = load_pair(vp, vp + 16);
      acc[nt] = __builtin_amdgcn_wmma_f32_16x16x32_bf16(false, a2, false, b2, (short)0, acc[nt], false, false);
    }
  }
  // done reading sh.p[w]; reuse the space as O-partial buffer
  #pragma unroll
  for (int nt = 0; nt < 4; nt++)
    #pragma unroll
    for (int j = 0; j < 8; j++)
      sh.o[w][j + half * 8][nt * 16 + lr] = acc[nt][j];
  __syncthreads();

  // ---- cross-wave reduce + store O ([h,b,t,dh]), non-temporal ----
  const int row = threadIdx.x >> 4;
  const int nb  = (threadIdx.x & 15) * 4;
  f4v s = {0.f, 0.f, 0.f, 0.f};
  #pragma unroll
  for (int ww = 0; ww < 8; ww++) {
    const float* pp = &sh.o[ww][row][nb];
    s.x += pp[0]; s.y += pp[1]; s.z += pp[2]; s.w += pp[3];
  }
  const size_t obase = ((size_t)(h * BATCH + b) * SEQ + qb * 16 + row) * (size_t)DH + nb;
  __builtin_nontemporal_store(s, (f4v*)(out + obase));
}

// ---------------------------------------------------------------------------
// Host launcher
// ---------------------------------------------------------------------------
extern "C" void kernel_launch(void* const* d_in, const int* in_sizes, int n_in,
                              void* d_out, int out_size, void* d_ws, size_t ws_size,
                              hipStream_t stream) {
  (void)in_sizes; (void)n_in; (void)out_size; (void)ws_size;
  // inputs: q, k (shape-only, unused per reference), v, pad_mask, W_q, b_q, W_k, b_k
  const float*         v    = (const float*)d_in[2];
  const unsigned char* mask = (const unsigned char*)d_in[3];
  const float*         Wq   = (const float*)d_in[4];
  const float*         bq   = (const float*)d_in[5];
  const float*         Wk   = (const float*)d_in[6];
  const float*         bk   = (const float*)d_in[7];

  float* out  = (float*)d_out;                                    // [h,b,t,dh]
  float* attn = out + (size_t)NHEAD * BATCH * SEQ * DH;           // [h,b,t,t]

  // workspace layout (bf16): Wq(256K) | Wk(256K) | Qh(8M) | Kh(8M) | Vt(16M)
  char* wsb = (char*)d_ws;
  bf16_t* Wqb = (bf16_t*)(wsb);
  bf16_t* Wkb = (bf16_t*)(wsb + 262144);
  bf16_t* Qh  = (bf16_t*)(wsb + 524288);
  bf16_t* Kh  = (bf16_t*)(wsb + 524288 + 8388608);
  bf16_t* Vt  = (bf16_t*)(wsb + 524288 + 16777216);

  mha_cvt_w   <<<512,   256, 0, stream>>>(Wq, Wk, Wqb, Wkb);
  mha_build_vt<<<32768, 256, 0, stream>>>(v, Vt);
  mha_proj    <<<dim3(1024, 16), 32, 0, stream>>>(v, Wqb, Wkb, bq, bk, Qh, Kh);
  mha_attn    <<<dim3(64, 128), 256, 0, stream>>>(Qh, Kh, Vt, mask, out, attn);
}
